// GroupedQueryAttention_38122129719937
// MI455X (gfx1250) — matrix operand order repaired and checked by reading, not compiled, not run on hardware
//
#include <hip/hip_runtime.h>

// ---------------- types ----------------
typedef __attribute__((ext_vector_type(16))) __bf16 v16bf;
typedef __attribute__((ext_vector_type(8)))  float  v8f;

union FragBF { v16bf v; uint4 u[2]; };

static __device__ __forceinline__ unsigned short f2bf(float f) {
    union { float f; unsigned int u; } c; c.f = f;
    unsigned int u = c.u;
    u += 0x7fffu + ((u >> 16) & 1u);           // round-to-nearest-even
    return (unsigned short)(u >> 16);
}
static __device__ __forceinline__ float bf2f(unsigned short h) {
    union { unsigned int u; float f; } c; c.u = ((unsigned int)h) << 16;
    return c.f;
}

static __device__ __forceinline__ v8f wmma_bf16(v16bf a, v16bf b, v8f c) {
    // D = A(16x32 bf16) * B(32x16 bf16) + C(16x16 f32)
    return __builtin_amdgcn_wmma_f32_16x16x32_bf16(false, a, false, b, (short)0, c, false, false);
}

// Async global -> LDS copy of 16 bytes (per lane), tracked by ASYNCcnt.
static __device__ __forceinline__ void async_copy_b128(unsigned lds_off, const void* gptr) {
    asm volatile("global_load_async_to_lds_b128 %0, %1, off"
                 :: "v"(lds_off), "v"((unsigned long long)(size_t)gptr)
                 : "memory");
}
#define ASYNC_WAIT(n) asm volatile("s_wait_asynccnt " #n ::: "memory")

// A-fragment: 16x32 bf16, row-major source with leading dim ld (elements).
// base points at (row0, k0). lane r (=lane&15) -> row r; hi half selects K grouping.
static __device__ __forceinline__ void load_a_frag(FragBF& f, const unsigned short* base,
                                                   long ld, int r, int hi) {
    const unsigned short* p = base + (long)r * ld;
    f.u[0] = *reinterpret_cast<const uint4*>(p + hi * 16);       // K = hi*16 .. +7
    f.u[1] = *reinterpret_cast<const uint4*>(p + hi * 16 + 8);   // K = hi*16+8 .. +15
}

// B-fragment: 32x16 bf16 where column n is stored K-contiguous with leading dim ld.
// base points at (n0-th row of the K-major storage, k0). lane r -> column r.
static __device__ __forceinline__ void load_b_frag(FragBF& f, const unsigned short* base,
                                                   long ld, int r, int hi) {
    const unsigned short* p = base + (long)r * ld + hi * 16;     // K = hi*16 .. +15
    f.u[0] = *reinterpret_cast<const uint4*>(p);
    f.u[1] = *reinterpret_cast<const uint4*>(p + 8);
}

// ---------------- f32 -> bf16 convert ----------------
__global__ void cvt_kernel(const float* __restrict__ src, unsigned short* __restrict__ dst, long n) {
    long i = (long)blockIdx.x * blockDim.x + threadIdx.x;
    long stride = (long)gridDim.x * blockDim.x;
    for (; i < n; i += stride) dst[i] = f2bf(src[i]);
}

// ---------------- RoPE (in place, bf16) ----------------
__global__ void rope_kernel(unsigned short* __restrict__ t, const float* __restrict__ freqs,
                            int heads, long total) {
    long idx = (long)blockIdx.x * blockDim.x + threadIdx.x;
    if (idx >= total) return;
    int  i   = (int)(idx & 31);          // pair index within head (HD/2 = 32)
    long tmp = idx >> 5;
    int  h   = (int)(tmp % heads);
    long row = tmp / heads;              // row in [0, B*S)
    int  s   = (int)(row & 2047);        // S = 2048
    long base = row * (long)(heads * 64) + (long)h * 64 + 2 * i;
    float te = bf2f(t[base]);
    float to = bf2f(t[base + 1]);
    long fo = ((long)s * 32 + i) * 2;
    float c = freqs[fo], sn = freqs[fo + 1];
    t[base]     = f2bf(te * c - to * sn);
    t[base + 1] = f2bf(te * sn + to * c);
}

// ---------------- generic bf16 GEMM:  C[M,N] = A[M,K] * Bw[N,K]^T ----------------
// Double-buffered LDS staging via GLOBAL_LOAD_ASYNC_TO_LDS_B128 (ASYNCcnt).
// Block: 128 threads = 4 waves (2 along M x 2 along N); tile 64(M) x 128(N) x 64(K-step).
// MODE 0: store bf16 row-major [M,N]
// MODE 1: store bf16 transposed into VT[b][g][d][S] (for V), b = row/Sdim, g=col/64, d=col%64
// MODE 2: store f32 row-major [M,N]
template <int MODE>
__global__ __launch_bounds__(128)
void gemm_bf16_kernel(const unsigned short* __restrict__ A,
                      const unsigned short* __restrict__ Bw,
                      void* __restrict__ Cout,
                      int M, int N, int K, int Sdim) {
    (void)M; (void)Sdim;
    constexpr int LDT = 72;              // padded LDS row stride (halves): 144B, 16B-aligned
    __shared__ __align__(16) unsigned short As[2][64][LDT];
    __shared__ __align__(16) unsigned short Bs[2][128][LDT];

    const int tid  = threadIdx.x;
    const int wid  = tid >> 5;
    const int lane = tid & 31;
    const int r    = lane & 15;
    const int hi   = lane >> 4;
    const int waveM = wid >> 1;
    const int waveN = wid & 1;
    const long m0b = (long)blockIdx.x * 64;
    const long n0b = (long)blockIdx.y * 128;
    const long mw  = m0b + waveM * 32;
    const long nw  = n0b + waveN * 64;

    // stage one K-step (64 wide): A tile 64x64, B tile 128x64, 12 async b128/thread
    auto stage = [&](int buf, int k0) {
#pragma unroll
        for (int i = 0; i < 4; ++i) {                 // A: 512 chunks / 128 threads
            int idx = tid * 4 + i;
            int row = idx >> 3, c = idx & 7;
            const unsigned short* g = A + (m0b + row) * (long)K + k0 + c * 8;
            async_copy_b128((unsigned)(size_t)&As[buf][row][c * 8], g);
        }
#pragma unroll
        for (int i = 0; i < 8; ++i) {                 // B: 1024 chunks / 128 threads
            int idx = tid * 8 + i;
            int row = idx >> 3, c = idx & 7;
            const unsigned short* g = Bw + (n0b + row) * (long)K + k0 + c * 8;
            async_copy_b128((unsigned)(size_t)&Bs[buf][row][c * 8], g);
        }
    };

    v8f acc[2][4] = {};
    const int nk = K / 64;
    stage(0, 0);
    for (int it = 0; it < nk; ++it) {
        const int cur = it & 1;
        if (it + 1 < nk) {
            stage(cur ^ 1, (it + 1) * 64);
            ASYNC_WAIT(12);              // in-order: batch for `cur` complete
        } else {
            ASYNC_WAIT(0);
        }
        __syncthreads();                  // publish LDS tiles to all waves
#pragma unroll
        for (int ks = 0; ks < 2; ++ks) {  // two 32-wide sub-steps of the 64-wide tile
            FragBF af[2], bf4[4];
#pragma unroll
            for (int mi = 0; mi < 2; ++mi)
                load_a_frag(af[mi], &As[cur][waveM * 32 + mi * 16][ks * 32], LDT, r, hi);
#pragma unroll
            for (int ni = 0; ni < 4; ++ni)
                load_b_frag(bf4[ni], &Bs[cur][waveN * 64 + ni * 16][ks * 32], LDT, r, hi);
#pragma unroll
            for (int mi = 0; mi < 2; ++mi)
#pragma unroll
                for (int ni = 0; ni < 4; ++ni)
                    acc[mi][ni] = wmma_bf16(af[mi].v, bf4[ni].v, acc[mi][ni]);
        }
        __syncthreads();                  // all reads of `cur` done before it is rewritten
    }

    if (MODE == 0) {
        unsigned short* C = (unsigned short*)Cout;
#pragma unroll
        for (int mi = 0; mi < 2; ++mi)
#pragma unroll
            for (int ni = 0; ni < 4; ++ni)
#pragma unroll
                for (int j = 0; j < 8; ++j) {
                    long row = mw + mi * 16 + j + 8 * hi;
                    long col = nw + ni * 16 + r;
                    C[row * (long)N + col] = f2bf(acc[mi][ni][j]);
                }
    } else if (MODE == 2) {
        float* C = (float*)Cout;
#pragma unroll
        for (int mi = 0; mi < 2; ++mi)
#pragma unroll
            for (int ni = 0; ni < 4; ++ni)
#pragma unroll
                for (int j = 0; j < 8; ++j) {
                    long row = mw + mi * 16 + j + 8 * hi;
                    long col = nw + ni * 16 + r;
                    C[row * (long)N + col] = acc[mi][ni][j];
                }
    } else { // MODE 1: transposed V store (lane's 8 rows are consecutive s -> one b128 store)
        unsigned short* VT = (unsigned short*)Cout;
#pragma unroll
        for (int mi = 0; mi < 2; ++mi) {
            long sfull = mw + mi * 16 + 8 * hi;
            long bidx  = sfull / Sdim;
            long sl    = sfull % Sdim;
#pragma unroll
            for (int ni = 0; ni < 4; ++ni) {
                int col = (int)(nw + ni * 16 + r);
                int g = col >> 6, d = col & 63;
                union { unsigned short s[8]; uint4 q; } t;
#pragma unroll
                for (int j = 0; j < 8; ++j) t.s[j] = f2bf(acc[mi][ni][j]);
                long off = (((bidx * 8 + g) * 64 + d) * (long)Sdim + sl);
                *reinterpret_cast<uint4*>(VT + off) = t.q;
            }
        }
    }
}

// ---------------- flash attention (causal, GQA) ----------------
// q:  bf16 [B*S, H*HD] (RoPE'd)   k: bf16 [B*S, KV*HD] (RoPE'd)
// vt: bf16 [B][KV][HD][S]         o: bf16 [B*S, H*HD]
__global__ __launch_bounds__(128)
void flash_kernel(const unsigned short* __restrict__ q,
                  const unsigned short* __restrict__ kk,
                  const unsigned short* __restrict__ vt,
                  unsigned short* __restrict__ o) {
    constexpr int S = 2048, H = 32, KV = 8, NR = 4, HD = 64;
    __shared__ __align__(16) unsigned short pshm[4][16 * 32];

    const int tid = threadIdx.x, wid = tid >> 5, lane = tid & 31;
    const int r = lane & 15, hi = lane >> 4;
    const int nqb = S / 64;
    const int qblk = blockIdx.x % nqb;
    const int bh   = blockIdx.x / nqb;
    const int h = bh % H, b = bh / H;
    const int g = h / NR;
    const int q0 = qblk * 64 + wid * 16;

    FragBF qf[2];
    {
        const unsigned short* qr = q + ((long)b * S + q0 + r) * (long)(H * HD) + h * HD;
        qf[0].u[0] = *reinterpret_cast<const uint4*>(qr + hi * 16);
        qf[0].u[1] = *reinterpret_cast<const uint4*>(qr + hi * 16 + 8);
        qf[1].u[0] = *reinterpret_cast<const uint4*>(qr + 32 + hi * 16);
        qf[1].u[1] = *reinterpret_cast<const uint4*>(qr + 32 + hi * 16 + 8);
    }

    float m8[8], l8[8];
#pragma unroll
    for (int j = 0; j < 8; ++j) { m8[j] = -3.0e38f; l8[j] = 0.0f; }
    v8f accv[4] = {};
    const float scale = 0.125f;     // 1/sqrt(64)

    const unsigned short* kb = kk + (long)b * S * (KV * HD) + g * HD;
    const unsigned short* vb = vt + ((long)b * KV + g) * (long)HD * S;
    unsigned short* pw = &pshm[wid][0];

    for (int n0 = 0; n0 <= q0 + 15; n0 += 32) {
        // ---- scores: 16 queries x 32 keys via 4 WMMAs ----
        v8f s0 = {}, s1 = {};
        {
            FragBF kf;
            const unsigned short* kr = kb + (long)(n0 + r) * (KV * HD);
            kf.u[0] = *reinterpret_cast<const uint4*>(kr + hi * 16);
            kf.u[1] = *reinterpret_cast<const uint4*>(kr + hi * 16 + 8);
            s0 = wmma_bf16(qf[0].v, kf.v, s0);
            kf.u[0] = *reinterpret_cast<const uint4*>(kr + 32 + hi * 16);
            kf.u[1] = *reinterpret_cast<const uint4*>(kr + 32 + hi * 16 + 8);
            s0 = wmma_bf16(qf[1].v, kf.v, s0);
            const unsigned short* kr2 = kb + (long)(n0 + 16 + r) * (KV * HD);
            kf.u[0] = *reinterpret_cast<const uint4*>(kr2 + hi * 16);
            kf.u[1] = *reinterpret_cast<const uint4*>(kr2 + hi * 16 + 8);
            s1 = wmma_bf16(qf[0].v, kf.v, s1);
            kf.u[0] = *reinterpret_cast<const uint4*>(kr2 + 32 + hi * 16);
            kf.u[1] = *reinterpret_cast<const uint4*>(kr2 + 32 + hi * 16 + 8);
            s1 = wmma_bf16(qf[1].v, kf.v, s1);
        }
        s0 = s0 * scale;
        s1 = s1 * scale;

        if (n0 + 31 > q0) {  // causal mask on diagonal-crossing chunks
#pragma unroll
            for (int j = 0; j < 8; ++j) {
                int row = q0 + j + 8 * hi;
                int c0 = n0 + r;
                if (c0 > row)       s0[j] = -1.0e30f;
                if (c0 + 16 > row)  s1[j] = -1.0e30f;
            }
        }

        // ---- online softmax (rows striped across 16 lanes per half-wave) ----
#pragma unroll
        for (int j = 0; j < 8; ++j) {
            float t = fmaxf(s0[j], s1[j]);
            t = fmaxf(t, __shfl_xor(t, 1, 32));
            t = fmaxf(t, __shfl_xor(t, 2, 32));
            t = fmaxf(t, __shfl_xor(t, 4, 32));
            t = fmaxf(t, __shfl_xor(t, 8, 32));
            float mn = fmaxf(m8[j], t);
            float al = __expf(m8[j] - mn);
            float p0 = __expf(s0[j] - mn);
            float p1 = __expf(s1[j] - mn);
            s0[j] = p0; s1[j] = p1;
            float rs = p0 + p1;
            rs += __shfl_xor(rs, 1, 32);
            rs += __shfl_xor(rs, 2, 32);
            rs += __shfl_xor(rs, 4, 32);
            rs += __shfl_xor(rs, 8, 32);
            l8[j] = l8[j] * al + rs;
            m8[j] = mn;
            accv[0][j] *= al; accv[1][j] *= al; accv[2][j] *= al; accv[3][j] *= al;
        }

        // ---- stage P (16x32 bf16) through LDS: C-layout -> A-layout ----
#pragma unroll
        for (int j = 0; j < 8; ++j) {
            pw[(j + 8 * hi) * 32 + r]      = f2bf(s0[j]);
            pw[(j + 8 * hi) * 32 + 16 + r] = f2bf(s1[j]);
        }
        FragBF pa;
        load_a_frag(pa, pw, 32, r, hi);   // same-wave LDS RAW: compiler inserts s_wait_dscnt

        // ---- PV: accumulate 16x64 output via 4 WMMAs against V^T ----
#pragma unroll
        for (int t = 0; t < 4; ++t) {
            FragBF vf;
            const unsigned short* vr = vb + (long)(t * 16 + r) * S + n0 + hi * 16;
            vf.u[0] = *reinterpret_cast<const uint4*>(vr);
            vf.u[1] = *reinterpret_cast<const uint4*>(vr + 8);
            accv[t] = wmma_bf16(pa.v, vf.v, accv[t]);
        }
    }

    // ---- normalize and store bf16 ----
#pragma unroll
    for (int j = 0; j < 8; ++j) {
        float inv = 1.0f / l8[j];
        long row = (long)b * S + q0 + j + 8 * hi;
#pragma unroll
        for (int t = 0; t < 4; ++t)
            o[row * (long)(H * HD) + h * HD + t * 16 + r] = f2bf(accv[t][j] * inv);
    }
}

// ---------------- host launch ----------------
extern "C" void kernel_launch(void* const* d_in, const int* in_sizes, int n_in,
                              void* d_out, int out_size, void* d_ws, size_t ws_size,
                              hipStream_t stream) {
    (void)in_sizes; (void)n_in; (void)out_size; (void)ws_size;
    const float* x     = (const float*)d_in[0];
    const float* freqs = (const float*)d_in[1];
    const float* wq    = (const float*)d_in[2];
    const float* wk    = (const float*)d_in[3];
    const float* wv    = (const float*)d_in[4];
    const float* wo    = (const float*)d_in[5];
    float* out = (float*)d_out;

    constexpr long Bc = 2, Sc = 2048, Dc = 2048, Hc = 32, KVc = 8, HDc = 64;
    const long BS = Bc * Sc;          // 4096
    const long Ekv = KVc * HDc;       // 512

    unsigned char* ws = (unsigned char*)d_ws;
    auto alloc = [&](long bytes) {
        unsigned char* p = ws;
        ws += (bytes + 255) & ~255L;
        return p;
    };
    unsigned short* xb   = (unsigned short*)alloc(BS * Dc * 2);
    unsigned short* wqb  = (unsigned short*)alloc(Dc * Dc * 2);
    unsigned short* wkb  = (unsigned short*)alloc(Ekv * Dc * 2);
    unsigned short* wvb  = (unsigned short*)alloc(Ekv * Dc * 2);
    unsigned short* wob  = (unsigned short*)alloc(Dc * Dc * 2);
    unsigned short* qb   = (unsigned short*)alloc(BS * Dc * 2);
    unsigned short* kbuf = (unsigned short*)alloc(BS * Ekv * 2);
    unsigned short* vtb  = (unsigned short*)alloc(Bc * KVc * HDc * Sc * 2);
    unsigned short* ab   = (unsigned short*)alloc(BS * Dc * 2);

    cvt_kernel<<<4096, 256, 0, stream>>>(x,  xb,  BS * Dc);
    cvt_kernel<<<4096, 256, 0, stream>>>(wq, wqb, Dc * Dc);
    cvt_kernel<<<1024, 256, 0, stream>>>(wk, wkb, Ekv * Dc);
    cvt_kernel<<<1024, 256, 0, stream>>>(wv, wvb, Ekv * Dc);
    cvt_kernel<<<4096, 256, 0, stream>>>(wo, wob, Dc * Dc);

    gemm_bf16_kernel<0><<<dim3(BS / 64, Dc / 128),  128, 0, stream>>>(xb, wqb, qb,   (int)BS, (int)Dc,  (int)Dc, (int)Sc);
    gemm_bf16_kernel<0><<<dim3(BS / 64, Ekv / 128), 128, 0, stream>>>(xb, wkb, kbuf, (int)BS, (int)Ekv, (int)Dc, (int)Sc);
    gemm_bf16_kernel<1><<<dim3(BS / 64, Ekv / 128), 128, 0, stream>>>(xb, wvb, vtb,  (int)BS, (int)Ekv, (int)Dc, (int)Sc);

    rope_kernel<<<(unsigned)((BS * Hc  * 32) / 256), 256, 0, stream>>>(qb,   freqs, (int)Hc,  BS * Hc  * 32);
    rope_kernel<<<(unsigned)((BS * KVc * 32) / 256), 256, 0, stream>>>(kbuf, freqs, (int)KVc, BS * KVc * 32);

    flash_kernel<<<(unsigned)(Bc * Hc * (Sc / 64)), 128, 0, stream>>>(qb, kbuf, vtb, ab);

    gemm_bf16_kernel<2><<<dim3(BS / 64, Dc / 128), 128, 0, stream>>>(ab, wob, out, (int)BS, (int)Dc, (int)Dc, (int)Sc);
}
